// SchNet_5909874999439
// MI455X (gfx1250) — matrix operand order repaired and checked by reading, not compiled, not run on hardware
//
#include <hip/hip_runtime.h>
#include <hip/hip_bf16.h>

// ---------------- problem constants ----------------
#define NATOMS 768
#define NEDGES (768 * 767)        // 589056
#define RTILES (NEDGES / 16)      // 36816 readout tiles (src-major edge order)
#define NJB    48                 // dst blocks of 16 atoms
#define CTILES (NJB * NATOMS)     // 36864 conv tiles: (jb, i) incl. dummy rows
#define NCENT  50
#define GAPF   (5.0f / 49.0f)     // linspace(0,5,50) spacing
#define INVGAP (49.0f / 5.0f)     // 1/gap
#define LOG2E  1.4426950408889634f
#define LN2    0.6931471805599453f
#define NGRP   16                 // src groups per dst block (48 src atoms each)

// fast hw transcendentals (arg to log is >= 1 here: no denorm fixup needed)
#if __has_builtin(__builtin_amdgcn_exp2f)
#define EXP2F(x) __builtin_amdgcn_exp2f(x)
#else
#define EXP2F(x) exp2f(x)
#endif
#if __has_builtin(__builtin_amdgcn_logf)
#define LOG2F(x) __builtin_amdgcn_logf(x)
#else
#define LOG2F(x) __log2f(x)
#endif

// ---------------- WMMA types ----------------
typedef __attribute__((ext_vector_type(16))) __bf16 v16bf;
typedef __attribute__((ext_vector_type(8)))  float  v8f;

union ABfrag { unsigned u[8]; uint4 q[2]; v16bf v; };

__device__ __forceinline__ unsigned f2bf_u(float f) {
    unsigned u = __float_as_uint(f);
    unsigned r = u + 0x7FFFu + ((u >> 16) & 1u);   // round-to-nearest-even
    return r >> 16;
}
__device__ __forceinline__ unsigned pack_bf2(float lo, float hi) {
    return f2bf_u(lo) | (f2bf_u(hi) << 16);
}
// Branchless softplus_bt(x, beta=0.5, thr=14): 2*log(1+exp(0.5x)).
// softplus(x) >= x always, so fmax(x, sp(min(x,28))) matches the reference.
__device__ __forceinline__ float softplus_b05(float x) {
    float t  = fminf(x, 28.0f);
    float e  = EXP2F(t * (0.5f * LOG2E));
    float sp = (2.0f * LN2) * LOG2F(1.0f + e);
    return fmaxf(x, sp);
}
// Branchless softplus_bt(x, beta=1, thr=20)
__device__ __forceinline__ float softplus_b1(float x) {
    float t = fminf(x, 20.0f);
    float e = EXP2F(t * LOG2E);
    return fmaxf(x, LN2 * LOG2F(1.0f + e));
}
__device__ __forceinline__ float rbf_val(float d, int k) {
    float x = d - (float)k * GAPF;
    return EXP2F((-INVGAP * LOG2E) * x * x);
}

// A-fragment K index for 16-bit A 16x32 layout (ISA 7.12.2); always even.
__device__ __forceinline__ int afrag_k(int v, int hh) {
    return (v < 4 ? v * 2 : 16 + (v - 4) * 2) + hh * 8;
}

// ---- CDNA5 async global->LDS copy (ASYNCcnt-tracked; VDST VGPR = LDS byte offset) ----
__device__ __forceinline__ void async_copy_b128(void* lds_dst, const void* gsrc) {
    unsigned           l = (unsigned)(size_t)lds_dst;       // low 32 bits = LDS offset
    unsigned long long g = (unsigned long long)(size_t)gsrc;
    asm volatile("global_load_async_to_lds_b128 %0, %1, off" :: "v"(l), "v"(g) : "memory");
}
__device__ __forceinline__ void wait_async0() {
    asm volatile("s_wait_asynccnt 0x0" ::: "memory");
}

// ---------------- weight repack: fp32 (KxN) -> bf16 B-fragments ----------------
// Per-lane-contiguous layout: dword = ((nt*2 + kf)*32 + lane)*8 + v
//   value = W[k][n], k = kf*32 + afrag_k(v, lane>>4)(+1), n = nt*16 + (lane&15)
__global__ void pack_weights_kernel(const float* __restrict__ pw1,
                                    const float* __restrict__ pw2,
                                    const float* __restrict__ row1,
                                    unsigned* __restrict__ pw1B,
                                    unsigned* __restrict__ pw2B,
                                    unsigned* __restrict__ roW1B) {
    int t = blockIdx.x * blockDim.x + threadIdx.x;
    if (t >= 7 * 2048) return;
    int mat  = t / 2048;
    int rem  = t % 2048;
    int v    = rem & 7;
    int lane = (rem >> 3) & 31;
    int kf   = (rem >> 8) & 1;
    int nt   = rem >> 9;
    int n    = nt * 16 + (lane & 15);
    int k    = kf * 32 + afrag_k(v, lane >> 4);
    const float* W; int Kreal; unsigned* dstp;
    if (mat < 3)      { W = pw1 + mat * 50 * 64;       Kreal = 50; dstp = pw1B + mat * 2048; }
    else if (mat < 6) { W = pw2 + (mat - 3) * 64 * 64; Kreal = 64; dstp = pw2B + (mat - 3) * 2048; }
    else              { W = row1;                      Kreal = 52; dstp = roW1B; }
    float lo = (k     < Kreal) ? W[k * 64 + n]       : 0.0f;
    float hi = (k + 1 < Kreal) ? W[(k + 1) * 64 + n] : 0.0f;
    dstp[rem] = pack_bf2(lo, hi);
}

// ---------------- rbf expansion -> bf16 A-fragments, dst-major tiling -------
// Tile T = jb*768 + i covers edges (src=i, dst=j=jb*16+m), m=0..15.
// Row with j==i is a dummy (rbf=0, masked in the accumulate).
// Per-lane-contiguous: dword = (T*2 + kf)*256 + lane*8 + v
__global__ void pack_rbfA_kernel(const float* __restrict__ ed, unsigned* __restrict__ rbfA) {
    unsigned t = blockIdx.x * blockDim.x + threadIdx.x;
    if (t >= (unsigned)CTILES * 512u) return;
    int v    = t & 7;
    int lane = (t >> 3) & 31;
    int kf   = (t >> 8) & 1;
    unsigned T = t >> 9;
    int i  = (int)(T % NATOMS);
    int jb = (int)(T / NATOMS);
    int m  = lane & 15, hh = lane >> 4;
    int k  = kf * 32 + afrag_k(v, hh);
    int j  = jb * 16 + m;
    float lo = 0.0f, hi = 0.0f;
    if (j != i) {
        int ei = i * 767 + (j > i ? j - 1 : j);   // src-major edge index
        float d = ed[ei];
        lo = (k     < NCENT) ? rbf_val(d, k)     : 0.0f;
        hi = (k + 1 < NCENT) ? rbf_val(d, k + 1) : 0.0f;
    }
    rbfA[t] = pack_bf2(lo, hi);
}

// ---------------- fused per-edge conv layer (WMMA hot loop) ----------------
// Block = (jb, grp): dst block jb (16 atoms), src group grp (48 atoms).
// Each wave handles 6 src atoms; msg accumulates in 32 registers per lane
// (the C layout is tile-invariant), then one LDS ds_add_f32 reduction and
// 1024 global atomics per block. No atomics in the inner loop.
__global__ __launch_bounds__(256)
void edge_conv_kernel(const uint4* __restrict__ rbfA,
                      const uint4* __restrict__ pw1B, const float* __restrict__ pb1,
                      const uint4* __restrict__ pw2B, const float* __restrict__ pb2,
                      const float* __restrict__ nw,
                      float* __restrict__ agg) {
    __shared__ uint4 wB[1024];          // [0..511]=pw1 frags, [512..1023]=pw2 (16 KB)
    __shared__ float yst[8][16 * 64];   // per-wave 16x64 f32 staging (32 KB)
    __shared__ float accs[16 * 64];     // dst-block accumulator (4 KB)
    const int tid = threadIdx.x;
    for (int t = tid; t < 512; t += 256) {
        async_copy_b128(&wB[t],       pw1B + t);
        async_copy_b128(&wB[512 + t], pw2B + t);
    }
    for (int t = tid; t < 1024; t += 256) accs[t] = 0.0f;
    wait_async0();
    __syncthreads();

    const int lane = tid & 31, wid = tid >> 5;
    const int r = lane & 15, hh = lane >> 4;
    const int jb    = blockIdx.x >> 4;       // 0..47 dst block
    const int grp   = blockIdx.x & 15;       // 0..15 src group
    const int jbase = jb * 16;

    float wacc[4][8];
#pragma unroll
    for (int nt = 0; nt < 4; ++nt)
#pragma unroll
        for (int v = 0; v < 8; ++v) wacc[nt][v] = 0.0f;

    for (int kk = 0; kk < 6; ++kk) {
        const int isrc = grp * 48 + kk * 8 + wid;          // src atom of this tile
        const size_t T = (size_t)jb * NATOMS + isrc;
        const uint4* ap = rbfA + T * 128;
        __builtin_prefetch(rbfA + (T + 8) * 128, 0, 0);
        ABfrag a0, a1;
        a0.q[0] = ap[lane * 2];      a0.q[1] = ap[lane * 2 + 1];
        a1.q[0] = ap[64 + lane * 2]; a1.q[1] = ap[64 + lane * 2 + 1];

        // GEMM1: y1 = softplus(rbf @ pw1 + pb1) -> LDS (C layout -> row-major)
#pragma unroll
        for (int nt = 0; nt < 4; ++nt) {
            float bias = pb1[nt * 16 + r];
            v8f acc;
#pragma unroll
            for (int v = 0; v < 8; ++v) acc[v] = bias;
            ABfrag b0, b1;
            const int base = nt * 128 + lane * 2;          // ((nt*2+kf)*32+lane)*2 uint4
            b0.q[0] = wB[base];      b0.q[1] = wB[base + 1];
            b1.q[0] = wB[base + 64]; b1.q[1] = wB[base + 65];
            acc = __builtin_amdgcn_wmma_f32_16x16x32_bf16(false, a0.v, false, b0.v, (short)0, acc, false, false);
            acc = __builtin_amdgcn_wmma_f32_16x16x32_bf16(false, a1.v, false, b1.v, (short)0, acc, false, false);
#pragma unroll
            for (int v = 0; v < 8; ++v)
                yst[wid][(v + 8 * hh) * 64 + nt * 16 + r] = softplus_b05(acc[v]);
        }

        // rebuild y1 as bf16 A fragments (float2 -> ds_load_b64; wave-local, DS in-order)
        ABfrag c0, c1;
        const float2* yp = (const float2*)&yst[wid][r * 64];
#pragma unroll
        for (int v = 0; v < 8; ++v) {
            int k = afrag_k(v, hh);
            float2 p0 = yp[k >> 1];
            float2 p1 = yp[(k + 32) >> 1];
            c0.u[v] = pack_bf2(p0.x, p0.y);
            c1.u[v] = pack_bf2(p1.x, p1.y);
        }

        // GEMM2: e = y1 @ pw2 + pb2 ; msg = e * nw[isrc] ; accumulate in registers
#pragma unroll
        for (int nt = 0; nt < 4; ++nt) {
            float bias = pb2[nt * 16 + r];
            v8f acc;
#pragma unroll
            for (int v = 0; v < 8; ++v) acc[v] = bias;
            ABfrag b0, b1;
            const int base = 512 + nt * 128 + lane * 2;
            b0.q[0] = wB[base];      b0.q[1] = wB[base + 1];
            b1.q[0] = wB[base + 64]; b1.q[1] = wB[base + 65];
            acc = __builtin_amdgcn_wmma_f32_16x16x32_bf16(false, c0.v, false, b0.v, (short)0, acc, false, false);
            acc = __builtin_amdgcn_wmma_f32_16x16x32_bf16(false, c1.v, false, b1.v, (short)0, acc, false, false);
            const float w = nw[isrc * 64 + nt * 16 + r];   // coalesced row of nw
#pragma unroll
            for (int v = 0; v < 8; ++v) {
                float val = (jbase + v + 8 * hh == isrc) ? 0.0f : acc[v] * w;  // mask j==i
                wacc[nt][v] += val;
            }
        }
    }

    // reduce the 8 waves' register accumulators into the dst-block LDS tile
#pragma unroll
    for (int nt = 0; nt < 4; ++nt)
#pragma unroll
        for (int v = 0; v < 8; ++v)
            atomicAdd(&accs[(v + 8 * hh) * 64 + nt * 16 + r], wacc[nt][v]);
    __syncthreads();

    // combine the 16 src groups via global atomics (1024 per block)
    for (int idx = tid; idx < 1024; idx += 256)
        atomicAdd(&agg[jb * 1024 + idx], accs[idx]);
}

// ---------------- readout (WMMA on [ha_src, ha_dst, rbf] features) ----------
__global__ __launch_bounds__(256)
void readout_kernel(const float* __restrict__ ed, const float* __restrict__ ha,
                    const int* __restrict__ src, const int* __restrict__ dst,
                    const uint4* __restrict__ roW1B, const float* __restrict__ rob1,
                    const float* __restrict__ row2, const float* __restrict__ rob2,
                    float* __restrict__ out) {
    __shared__ uint4 wB[512];           // ro_w1 fragments (8 KB)
    __shared__ float fst[8][16 * 64];   // per-wave feature/y staging (32 KB)
    const int tid = threadIdx.x;
    for (int t = tid; t < 512; t += 256) async_copy_b128(&wB[t], roW1B + t);
    wait_async0();
    __syncthreads();

    const int lane = tid & 31, wid = tid >> 5;
    const int r = lane & 15, hh = lane >> 4;
    const int stride = gridDim.x * 8;

    for (int tile = blockIdx.x * 8 + wid; tile < RTILES; tile += stride) {
        const int ebase = tile * 16;
        // build 16x64 feature tile (K padded 52->64) in wave-local LDS
        for (int idx = lane; idx < 1024; idx += 32) {
            int m = idx >> 6, k = idx & 63;
            int ei = ebase + m;
            float v;
            if (k == 0)      v = ha[src[ei]];
            else if (k == 1) v = ha[dst[ei]];
            else if (k < 52) v = rbf_val(ed[ei], k - 2);
            else             v = 0.0f;
            fst[wid][idx] = v;
        }

        ABfrag a0, a1;
        const float2* fp = (const float2*)&fst[wid][r * 64];
#pragma unroll
        for (int v = 0; v < 8; ++v) {
            int k = afrag_k(v, hh);
            float2 p0 = fp[k >> 1];
            float2 p1 = fp[(k + 32) >> 1];
            a0.u[v] = pack_bf2(p0.x, p0.y);
            a1.u[v] = pack_bf2(p1.x, p1.y);
        }

        // y = relu(feat @ ro_w1 + ro_b1) -> back to LDS
#pragma unroll
        for (int nt = 0; nt < 4; ++nt) {
            float bias = rob1[nt * 16 + r];
            v8f acc;
#pragma unroll
            for (int v = 0; v < 8; ++v) acc[v] = bias;
            ABfrag b0, b1;
            const int base = nt * 128 + lane * 2;
            b0.q[0] = wB[base];      b0.q[1] = wB[base + 1];
            b1.q[0] = wB[base + 64]; b1.q[1] = wB[base + 65];
            acc = __builtin_amdgcn_wmma_f32_16x16x32_bf16(false, a0.v, false, b0.v, (short)0, acc, false, false);
            acc = __builtin_amdgcn_wmma_f32_16x16x32_bf16(false, a1.v, false, b1.v, (short)0, acc, false, false);
#pragma unroll
            for (int v = 0; v < 8; ++v)
                fst[wid][(v + 8 * hh) * 64 + nt * 16 + r] = fmaxf(acc[v], 0.0f);
        }

        // tiny 64x2 head + 2-way softmax, one edge per lane (lanes 0..15)
        if (lane < 16) {
            int ei = ebase + lane;
            float l0 = rob2[0], l1 = rob2[1];
            for (int n = 0; n < 64; ++n) {
                float y = fst[wid][lane * 64 + n];
                l0 += y * row2[n * 2 + 0];
                l1 += y * row2[n * 2 + 1];
            }
            float mx = fmaxf(l0, l1);
            float e0 = EXP2F((l0 - mx) * LOG2E), e1 = EXP2F((l1 - mx) * LOG2E);
            float s = e0 + e1;
            out[ei * 2 + 0] = e0 / s;
            out[ei * 2 + 1] = e1 / s;
        }
    }
}

// ---------------- small node-level kernels (negligible FLOPs) ----------------
__global__ void embed_kernel(const int* __restrict__ at, const float* __restrict__ emb,
                             float* __restrict__ h) {
    int i = blockIdx.x, n = threadIdx.x;
    h[i * 64 + n] = emb[at[i] * 64 + n];
}
__global__ void nw_zero_kernel(const float* __restrict__ h, const float* __restrict__ w1l,
                               float* __restrict__ nw, float* __restrict__ agg) {
    int i = blockIdx.x, n = threadIdx.x;
    float s = 0.0f;
    for (int k = 0; k < 64; ++k) s += h[i * 64 + k] * w1l[k * 64 + n];
    nw[i * 64 + n]  = s;
    agg[i * 64 + n] = 0.0f;   // zero accumulator for the scatter
}
__global__ void upd1_kernel(const float* __restrict__ agg, const float* __restrict__ qw1l,
                            const float* __restrict__ qb1l, float* __restrict__ tb) {
    int i = blockIdx.x, n = threadIdx.x;
    float s = qb1l[n];
    for (int k = 0; k < 64; ++k) s += agg[i * 64 + k] * qw1l[k * 64 + n];
    tb[i * 64 + n] = softplus_b05(s);
}
__global__ void upd2_kernel(const float* __restrict__ tb, const float* __restrict__ qw2l,
                            const float* __restrict__ qb2l, float* __restrict__ h) {
    int i = blockIdx.x, n = threadIdx.x;
    float s = qb2l[n];
    for (int k = 0; k < 64; ++k) s += tb[i * 64 + k] * qw2l[k * 64 + n];
    h[i * 64 + n] += s;
}
__global__ void atom1_kernel(const float* __restrict__ h, const float* __restrict__ auw1,
                             const float* __restrict__ aub1, float* __restrict__ t2) {
    int i = blockIdx.x, n = threadIdx.x;
    float s = aub1[n];
    for (int k = 0; k < 64; ++k) s += h[i * 64 + k] * auw1[k * 64 + n];
    t2[i * 64 + n] = softplus_b1(s) - LN2;   // ShiftSoftplus
}
__global__ void atom2_kernel(const float* __restrict__ t2, const float* __restrict__ auw2,
                             const float* __restrict__ aub2, float* __restrict__ ha) {
    int i = blockIdx.x * blockDim.x + threadIdx.x;
    if (i >= NATOMS) return;
    float s = aub2[0];
    for (int n = 0; n < 64; ++n) s += t2[i * 64 + n] * auw2[n];
    ha[i] = s;
}

// ---------------- launch ----------------
extern "C" void kernel_launch(void* const* d_in, const int* in_sizes, int n_in,
                              void* d_out, int out_size, void* d_ws, size_t ws_size,
                              hipStream_t stream) {
    (void)in_sizes; (void)n_in; (void)out_size; (void)ws_size;
    const int*   atom_types = (const int*)d_in[0];
    const float* ed   = (const float*)d_in[1];
    const int*   src  = (const int*)d_in[2];
    const int*   dst  = (const int*)d_in[3];
    const float* emb  = (const float*)d_in[4];
    const float* w1   = (const float*)d_in[5];
    const float* pw1  = (const float*)d_in[6];
    const float* pb1  = (const float*)d_in[7];
    const float* pw2  = (const float*)d_in[8];
    const float* pb2  = (const float*)d_in[9];
    const float* qw1  = (const float*)d_in[10];
    const float* qb1  = (const float*)d_in[11];
    const float* qw2  = (const float*)d_in[12];
    const float* qb2  = (const float*)d_in[13];
    const float* auw1 = (const float*)d_in[14];
    const float* aub1 = (const float*)d_in[15];
    const float* auw2 = (const float*)d_in[16];
    const float* aub2 = (const float*)d_in[17];
    const float* row1 = (const float*)d_in[18];
    const float* rob1 = (const float*)d_in[19];
    const float* row2 = (const float*)d_in[20];
    const float* rob2 = (const float*)d_in[21];
    float* out = (float*)d_out;

    // workspace layout (~76.5 MB total; rbfA is L2-resident and reused 3x)
    char* ws = (char*)d_ws;
    const size_t NM = (size_t)NATOMS * 64 * sizeof(float);   // 196608
    float*    h     = (float*)(ws + 0 * NM);
    float*    nw    = (float*)(ws + 1 * NM);
    float*    agg   = (float*)(ws + 2 * NM);
    float*    tb    = (float*)(ws + 3 * NM);
    float*    t2    = (float*)(ws + 4 * NM);
    float*    ha    = (float*)(ws + 5 * NM);                 // 768 floats (+pad)
    unsigned* pw1B  = (unsigned*)(ws + 5 * NM + 4096);
    unsigned* pw2B  = pw1B + 3 * 2048;
    unsigned* roW1B = pw2B + 3 * 2048;
    unsigned* rbfA  = roW1B + 2048;                          // 36864*512 dwords = 75.5 MB

    pack_weights_kernel<<<(7 * 2048 + 255) / 256, 256, 0, stream>>>(pw1, pw2, row1,
                                                                    pw1B, pw2B, roW1B);
    pack_rbfA_kernel<<<((unsigned)CTILES * 512u + 255u) / 256u, 256, 0, stream>>>(ed, rbfA);
    embed_kernel<<<NATOMS, 64, 0, stream>>>(atom_types, emb, h);

    for (int l = 0; l < 3; ++l) {
        nw_zero_kernel<<<NATOMS, 64, 0, stream>>>(h, w1 + l * 4096, nw, agg);
        edge_conv_kernel<<<NJB * NGRP, 256, 0, stream>>>((const uint4*)rbfA,
                                                         (const uint4*)(pw1B + l * 2048), pb1 + l * 64,
                                                         (const uint4*)(pw2B + l * 2048), pb2 + l * 64,
                                                         nw, agg);
        upd1_kernel<<<NATOMS, 64, 0, stream>>>(agg, qw1 + l * 4096, qb1 + l * 64, tb);
        upd2_kernel<<<NATOMS, 64, 0, stream>>>(tb, qw2 + l * 4096, qb2 + l * 64, h);
    }

    atom1_kernel<<<NATOMS, 64, 0, stream>>>(h, auw1, aub1, t2);
    atom2_kernel<<<3, 256, 0, stream>>>(t2, auw2, aub2, ha);
    readout_kernel<<<768, 256, 0, stream>>>(ed, ha, src, dst,
                                            (const uint4*)roW1B, rob1, row2, rob2, out);
}
